// AttLocRec_87832081203817
// MI455X (gfx1250) — matrix operand order repaired and checked by reading, not compiled, not run on hardware
//
#include <hip/hip_runtime.h>
#include <cstdint>
#include <cstddef>

#define B_      32
#define T_      2048
#define EPROJS_ 1024
#define ATT_    512
#define CCH_    10
#define KSZ_    201
#define PAD_    100

// Rows per block / per-wave tiling for the score GEMM
#define BM      64     // rows per block (4 M-tiles)
#define MT      4      // M-tiles per wave
#define NT      4      // N-tiles per wave

// LDS row stride (in bf16 elems): 1024 + 8 keeps 16B alignment of b128 loads
// and rotates each row by 4 banks -> conflict-free ds_load_b128.
#define LDSTRIDE 1032

typedef __attribute__((ext_vector_type(16))) __bf16          v16bf;
typedef __attribute__((ext_vector_type(8)))  unsigned int    v8u;
typedef __attribute__((ext_vector_type(16))) unsigned short  v16u;
typedef __attribute__((ext_vector_type(8)))  float           v8f;

#if defined(__has_builtin)
#if __has_builtin(__builtin_amdgcn_tanhf)
#define FAST_TANH(x) __builtin_amdgcn_tanhf(x)
#endif
#endif
#ifndef FAST_TANH
#define FAST_TANH(x) tanhf(x)
#endif

static __device__ __forceinline__ unsigned short f2b(float x) {
  // fp32 -> bf16 round-to-nearest-even
  unsigned u = __builtin_bit_cast(unsigned, x);
  unsigned r = 0x7FFFu + ((u >> 16) & 1u);
  return (unsigned short)((u + r) >> 16);
}
static __device__ __forceinline__ float sigmoidf_(float x) {
  return 1.0f / (1.0f + expf(-x));
}

// ---------------------------------------------------------------------------
// Pack W_enc (1024x512 f32, row-major K x N) into WMMA B-fragment order bf16:
// Bpack[((nt*32 + ks)*32 + lane)*16 + i]   (32 n-tiles, 32 k-steps of 32)
//   lane 0-15 : N = nt*16 + lane,      K = ks*32 + i        (i = 0..15)
//   lane 16-31: N = nt*16 + (lane-16), K = ks*32 + 16 + i
// Also zeroes the context output (accumulated by atomics each launch).
// ---------------------------------------------------------------------------
__global__ void pack_b_zero_c(const float* __restrict__ W_enc,
                              unsigned short* __restrict__ Bpack,
                              float* __restrict__ c_out) {
  int idx  = blockIdx.x * blockDim.x + threadIdx.x;   // 0 .. 524287
  int i    = idx & 15;
  int lane = (idx >> 4) & 31;
  int ks   = (idx >> 9) & 31;
  int nt   = idx >> 14;
  int n = nt * 16 + (lane & 15);
  int k = ks * 32 + ((lane >> 4) << 4) + i;
  Bpack[idx] = f2b(W_enc[k * ATT_ + n]);
  if (idx < B_ * EPROJS_) c_out[idx] = 0.0f;
}

// ---------------------------------------------------------------------------
// conv1d(att_prev, W_conv, pad=100) -> relu -> max over T  => feat (B, 10)
// ---------------------------------------------------------------------------
__global__ void conv_feat_kernel(const float* __restrict__ att_prev,
                                 const float* __restrict__ W_conv,
                                 float* __restrict__ feat) {
  __shared__ float row[T_ + 2 * PAD_];
  __shared__ float wk[KSZ_];
  __shared__ float red[256];
  int b = blockIdx.x, c = blockIdx.y, tid = threadIdx.x;
  for (int i = tid; i < T_ + 2 * PAD_; i += 256) {
    int t = i - PAD_;
    row[i] = (t >= 0 && t < T_) ? att_prev[b * T_ + t] : 0.0f;
  }
  for (int i = tid; i < KSZ_; i += 256) wk[i] = W_conv[c * KSZ_ + i];
  __syncthreads();
  float mx = 0.0f;  // relu => max >= 0
  for (int t = tid; t < T_; t += 256) {
    float s = 0.0f;
    for (int k = 0; k < KSZ_; ++k) s = fmaf(row[t + k], wk[k], s);
    mx = fmaxf(mx, s);
  }
  red[tid] = mx; __syncthreads();
  for (int s = 128; s > 0; s >>= 1) {
    if (tid < s) red[tid] = fmaxf(red[tid], red[tid + s]);
    __syncthreads();
  }
  if (tid == 0) feat[b * CCH_ + c] = red[0];
}

// ---------------------------------------------------------------------------
// LSTMCell (no bias), PyTorch gate order i,f,g,o.  Writes h_new, c_new.
// ---------------------------------------------------------------------------
__global__ void lstm_kernel(const float* __restrict__ feat,
                            const float* __restrict__ att_h,
                            const float* __restrict__ att_c,
                            const float* __restrict__ W_ih,
                            const float* __restrict__ W_hh,
                            float* __restrict__ h_out,
                            float* __restrict__ c_out) {
  __shared__ float gates[4 * ATT_];
  __shared__ float hsh[ATT_];
  __shared__ float fsh[CCH_];
  int b = blockIdx.x, tid = threadIdx.x;
  for (int i = tid; i < ATT_; i += 256) hsh[i] = att_h[b * ATT_ + i];
  if (tid < CCH_) fsh[tid] = feat[b * CCH_ + tid];
  __syncthreads();
  for (int j = tid; j < 4 * ATT_; j += 256) {
    float g = 0.0f;
    for (int c = 0; c < CCH_; ++c) g = fmaf(fsh[c], W_ih[j * CCH_ + c], g);
    for (int a = 0; a < ATT_; ++a) g = fmaf(hsh[a], W_hh[j * ATT_ + a], g);
    gates[j] = g;
  }
  __syncthreads();
  for (int a = tid; a < ATT_; a += 256) {
    float gi = gates[a], gf = gates[ATT_ + a], gg = gates[2 * ATT_ + a], go = gates[3 * ATT_ + a];
    float cn = sigmoidf_(gf) * att_c[b * ATT_ + a] + sigmoidf_(gi) * FAST_TANH(gg);
    float hn = sigmoidf_(go) * FAST_TANH(cn);
    c_out[b * ATT_ + a] = cn;
    h_out[b * ATT_ + a] = hn;
  }
}

// ---------------------------------------------------------------------------
// rbias[b,a] = b_enc[a] + h_new[b,a] + (dec_z @ W_dec)[b,a]
// ---------------------------------------------------------------------------
__global__ void rbias_kernel(const float* __restrict__ dec_z,
                             const float* __restrict__ W_dec,
                             const float* __restrict__ b_enc,
                             const float* __restrict__ h_new,
                             float* __restrict__ rbias) {
  __shared__ float z[EPROJS_];
  int b = blockIdx.x, tid = threadIdx.x;
  for (int i = tid; i < EPROJS_; i += 256) z[i] = dec_z[b * EPROJS_ + i];
  __syncthreads();
  for (int a = tid; a < ATT_; a += 256) {
    float s = b_enc[a] + h_new[b * ATT_ + a];
    for (int e = 0; e < EPROJS_; ++e) s = fmaf(z[e], W_dec[e * ATT_ + a], s);
    rbias[b * ATT_ + a] = s;
  }
}

// ---------------------------------------------------------------------------
// Fused score GEMM:  e[b,t] = b_g + sum_n W_g[n]*tanh(rbias[b,n] + (enc @ W_enc)[b,t,n])
// One block = 64 rows of (B*T), staged once as bf16 in LDS (~129 KB of the
// WGP's 320 KB).  Each of the 8 waves computes 4 M-tiles x 4 N-tiles: per
// k-step 8 ds_load_b128 + 8 global_load_b128 feed 16 WMMAs (1 load / WMMA),
// and each B fragment is reused by 4 independent WMMAs.  Four persistent
// per-M-tile LDS pointers are bumped once per 4-k-step group so all in-loop
// ds accesses use small immediate offsets (no in-loop address VALU on
// fragment registers, no WMMA WAR v_nop padding).
// ---------------------------------------------------------------------------
__global__ void __launch_bounds__(256)
score_kernel(const float* __restrict__ enc,
             const unsigned short* __restrict__ Bpack,
             const float* __restrict__ rbias,
             const float* __restrict__ W_g,
             const float* __restrict__ b_g,
             float* __restrict__ e_out) {
  __shared__ unsigned short lds_a[BM * LDSTRIDE];   // ~129 KB bf16 A tile
  __shared__ float lds_red[BM];
  const int tid  = threadIdx.x;
  const int wave = tid >> 5;
  const int lane = tid & 31;
  const int row0 = blockIdx.x * BM;      // global row in (B*T)
  const int b    = row0 >> 11;           // T = 2048 (BM divides T)
  const int m    = lane & 15;            // fragment row/col within tile
  const int kh   = lane >> 4;            // which K-half this lane holds

  if (tid < BM) lds_red[tid] = 0.0f;

  // ---- cooperative stage: fp32 (64x1024) -> bf16 LDS, coalesced float4 ----
  {
    const float4* src4 = (const float4*)(enc + (size_t)row0 * (size_t)EPROJS_);
#pragma unroll 4
    for (int it = 0; it < BM * 4; ++it) {   // BM*4 iters x 256 threads = 4x the BM*1024/4 float4s
      int g  = it * 256 + tid;
      int r  = g >> 8;                      // row (256 float4 per row)
      int c4 = g & 255;                     // float4 index within row
      float4 v = src4[g];
      unsigned lo = (unsigned)f2b(v.x) | ((unsigned)f2b(v.y) << 16);
      unsigned hi = (unsigned)f2b(v.z) | ((unsigned)f2b(v.w) << 16);
      unsigned* p = (unsigned*)(lds_a + r * LDSTRIDE + c4 * 4);
      p[0] = lo; p[1] = hi;
    }
  }
  __syncthreads();

  v8f acc[MT * NT];
#pragma unroll
  for (int i = 0; i < MT * NT; ++i) acc[i] = (v8f){};

  const int nb = wave * NT;              // first of 4 n-tiles for this wave
  const unsigned short* bp = Bpack + ((size_t)nb * 32 * 32 + lane) * 16;

  // Persistent per-M-tile LDS pointers, bumped once per 4-k-step group.
  const unsigned short* ap0 = lds_a + m * LDSTRIDE + kh * 16;
  const unsigned short* ap1 = ap0 + 16 * LDSTRIDE;
  const unsigned short* ap2 = ap0 + 32 * LDSTRIDE;
  const unsigned short* ap3 = ap0 + 48 * LDSTRIDE;

  for (int ksg = 0; ksg < 8; ++ksg) {    // 8 groups x 4 k-steps
#pragma unroll
    for (int kk = 0; kk < 4; ++kk) {
      const int ks = ksg * 4 + kk;       // B index (runtime ksg folds into addr)
      // A fragments for the 4 M-tiles (two b128 LDS loads each, imm offsets)
      v16bf afr[MT];
#pragma unroll
      for (int mt = 0; mt < MT; ++mt) {
        const unsigned short* ap = (mt == 0) ? ap0 : (mt == 1) ? ap1
                                 : (mt == 2) ? ap2 : ap3;
        uint4 lo = *(const uint4*)(ap + kk * 32);
        uint4 hi = *(const uint4*)(ap + kk * 32 + 8);
        v8u af;
        af[0] = lo.x; af[1] = lo.y; af[2] = lo.z; af[3] = lo.w;
        af[4] = hi.x; af[5] = hi.y; af[6] = hi.z; af[7] = hi.w;
        afr[mt] = __builtin_bit_cast(v16bf, af);
      }
      // 4 B fragments, each feeding 4 independent WMMAs
#pragma unroll
      for (int nt = 0; nt < NT; ++nt) {
        v16u bu = *(const v16u*)(bp + (size_t)ks * 512 + (size_t)nt * 16384);
        v16bf bb = __builtin_bit_cast(v16bf, bu);
#pragma unroll
        for (int mt = 0; mt < MT; ++mt) {
          acc[mt * NT + nt] = __builtin_amdgcn_wmma_f32_16x16x32_bf16(
              false, afr[mt], false, bb, (short)0, acc[mt * NT + nt], false, false);
        }
      }
    }
    ap0 += 128; ap1 += 128; ap2 += 128; ap3 += 128;   // 4 k-steps * 32 elems
  }

  // Epilogue: tanh(acc + rbias[col]) * W_g[col], reduced over this wave's cols.
  // C layout: lane holds col = ntile*16 + m, rows r + 8*kh in VGPR r.
  float ps[MT][8];
#pragma unroll
  for (int mt = 0; mt < MT; ++mt)
#pragma unroll
    for (int r = 0; r < 8; ++r) ps[mt][r] = 0.0f;

#pragma unroll
  for (int nt = 0; nt < NT; ++nt) {
    int col = (nb + nt) * 16 + m;
    float rb = rbias[b * ATT_ + col];
    float wg = W_g[col];
#pragma unroll
    for (int mt = 0; mt < MT; ++mt)
#pragma unroll
      for (int r = 0; r < 8; ++r)
        ps[mt][r] += FAST_TANH(acc[mt * NT + nt][r] + rb) * wg;
  }
  // butterfly reduce across 16 lanes holding the same rows (masks <= 8 stay in-half)
#pragma unroll
  for (int mt = 0; mt < MT; ++mt) {
#pragma unroll
    for (int r = 0; r < 8; ++r) {
      float v = ps[mt][r];
      v += __shfl_xor(v, 1);
      v += __shfl_xor(v, 2);
      v += __shfl_xor(v, 4);
      v += __shfl_xor(v, 8);
      ps[mt][r] = v;
    }
  }
  if (m == 0) {
#pragma unroll
    for (int mt = 0; mt < MT; ++mt)
#pragma unroll
      for (int r = 0; r < 8; ++r)
        atomicAdd(&lds_red[mt * 16 + kh * 8 + r], ps[mt][r]);
  }
  __syncthreads();
  if (tid < BM) e_out[row0 + tid] = lds_red[tid] + b_g[0];
}

// ---------------------------------------------------------------------------
// Masked softmax over T with SCALING=2:  w = softmax(2*e) over t < len, else 0.
// ---------------------------------------------------------------------------
__global__ void softmax_kernel(const float* __restrict__ e,
                               const int* __restrict__ lens,
                               float* __restrict__ w_out) {
  __shared__ float red[256];
  int b = blockIdx.x, tid = threadIdx.x;
  int len = lens[b];
  const float* eb = e + b * T_;
  float mx = -3.4e38f;
  for (int t = tid; t < T_; t += 256)
    if (t < len) mx = fmaxf(mx, 2.0f * eb[t]);
  red[tid] = mx; __syncthreads();
  for (int s = 128; s > 0; s >>= 1) {
    if (tid < s) red[tid] = fmaxf(red[tid], red[tid + s]);
    __syncthreads();
  }
  mx = red[0]; __syncthreads();
  float sum = 0.0f;
  for (int t = tid; t < T_; t += 256)
    if (t < len) sum += expf(2.0f * eb[t] - mx);
  red[tid] = sum; __syncthreads();
  for (int s = 128; s > 0; s >>= 1) {
    if (tid < s) red[tid] += red[tid + s];
    __syncthreads();
  }
  float inv = 1.0f / red[0];
  for (int t = tid; t < T_; t += 256)
    w_out[b * T_ + t] = (t < len) ? expf(2.0f * eb[t] - mx) * inv : 0.0f;
}

// ---------------------------------------------------------------------------
// Context: c[b,:] = sum_t w[b,t] * enc[b,t,:].  Grid (B, 4 col-chunks, 8 t-chunks),
// partial sums merged with global f32 atomics (c zeroed in pack_b_zero_c).
// ---------------------------------------------------------------------------
__global__ void context_kernel(const float* __restrict__ w,
                               const float* __restrict__ enc,
                               float* __restrict__ c) {
  __shared__ float wsh[256];
  int b = blockIdx.x, cc = blockIdx.y, tc = blockIdx.z, tid = threadIdx.x;
  wsh[tid] = w[b * T_ + tc * 256 + tid];
  __syncthreads();
  int col = cc * 256 + tid;
  const float* base = enc + ((size_t)b * T_ + (size_t)tc * 256) * EPROJS_ + col;
  float acc = 0.0f;
  for (int t = 0; t < 256; ++t) acc = fmaf(wsh[t], base[(size_t)t * EPROJS_], acc);
  atomicAdd(&c[b * EPROJS_ + col], acc);
}

// ---------------------------------------------------------------------------
extern "C" void kernel_launch(void* const* d_in, const int* in_sizes, int n_in,
                              void* d_out, int out_size, void* d_ws, size_t ws_size,
                              hipStream_t stream) {
  const float* enc      = (const float*)d_in[0];
  const int*   lens     = (const int*)d_in[1];
  const float* dec_z    = (const float*)d_in[2];
  const float* att_prev = (const float*)d_in[3];
  const float* att_h    = (const float*)d_in[4];
  const float* att_c    = (const float*)d_in[5];
  const float* W_enc    = (const float*)d_in[6];
  const float* b_enc    = (const float*)d_in[7];
  const float* W_dec    = (const float*)d_in[8];
  const float* W_conv   = (const float*)d_in[9];
  const float* W_ih     = (const float*)d_in[10];
  const float* W_hh     = (const float*)d_in[11];
  const float* W_g      = (const float*)d_in[12];
  const float* b_g      = (const float*)d_in[13];

  float* out    = (float*)d_out;
  float* c_out  = out;                       // (B, EPROJS)  32768
  float* w_out  = out + B_ * EPROJS_;        // (B, T)       65536
  float* h_out  = w_out + B_ * T_;           // (B, ATT)     16384
  float* cn_out = h_out + B_ * ATT_;         // (B, ATT)     16384

  char* ws = (char*)d_ws;
  unsigned short* Bpack = (unsigned short*)ws;                     // 1 MB
  float* feat  = (float*)(ws + (1u << 20));                        // 320 f
  float* rbias = (float*)(ws + (1u << 20) + 4096);                 // 64 KB
  float* e_buf = (float*)(ws + (1u << 20) + 4096 + 65536);         // 256 KB

  pack_b_zero_c<<<2048, 256, 0, stream>>>(W_enc, Bpack, c_out);
  conv_feat_kernel<<<dim3(B_, CCH_), 256, 0, stream>>>(att_prev, W_conv, feat);
  lstm_kernel<<<B_, 256, 0, stream>>>(feat, att_h, att_c, W_ih, W_hh, h_out, cn_out);
  rbias_kernel<<<B_, 256, 0, stream>>>(dec_z, W_dec, b_enc, h_out, rbias);
  score_kernel<<<(B_ * T_) / BM, 256, 0, stream>>>(enc, Bpack, rbias, W_g, b_g, e_buf);
  softmax_kernel<<<B_, 256, 0, stream>>>(e_buf, lens, w_out);
  context_kernel<<<dim3(B_, 4, 8), 256, 0, stream>>>(w_out, enc, c_out);
}
